// MLPModel_59665685676275
// MI455X (gfx1250) — matrix operand order repaired and checked by reading, not compiled, not run on hardware
//
#include <hip/hip_runtime.h>

// ---------- types ----------
typedef __attribute__((ext_vector_type(16))) __bf16         v16bf;
typedef __attribute__((ext_vector_type(2)))  __bf16         v2bf;
typedef __attribute__((ext_vector_type(8)))  float          v8f;
typedef __attribute__((ext_vector_type(8)))  unsigned short u16x8;
typedef __attribute__((ext_vector_type(4)))  unsigned int   tdm_g0_t;
typedef __attribute__((ext_vector_type(8)))  int            tdm_g1_t;
typedef __attribute__((ext_vector_type(4)))  int            tdm_g2_t;

union Frag {
    u16x8 h[2];
    v16bf bf;
};

// pack two fp32 -> one dword of two bf16 (lo = a, hi = b), RNE
__device__ inline unsigned pack_bf16x2(float a, float b) {
#if __has_builtin(__builtin_amdgcn_cvt_pk_bf16_f32)
    union { v2bf v; unsigned u; } c;
    c.v = __builtin_amdgcn_cvt_pk_bf16_f32(a, b);   // v_cvt_pk_bf16_f32
    return c.u;
#else
    unsigned r;
    asm("v_cvt_pk_bf16_f32 %0, %1, %2" : "=v"(r) : "v"(a), "v"(b));
    return r;
#endif
}
__device__ inline unsigned short f32_to_bf16(float f) {
    return (unsigned short)(pack_bf16x2(f, f) & 0xffffu);
}

// ---------- tiling ----------
constexpr int TBM = 128;      // block tile rows (batch)
constexpr int TBN = 64;       // block tile cols (output features)
constexpr int TK  = 32;       // K step (bf16 wmma depth)
constexpr int LDA = TK + 8;   // padded LDS stride in ushorts (rows stay 16B aligned)
constexpr int LDW = TK + 8;   // 32 data ushorts (16 DW) + 8 pad ushorts (4 DW)

// ---------- TDM: 2-D bf16 tile load, LDS padding = 4 DWORDs every 16 DWORDs ----------
// Descriptor per CDNA5 ISA section 8.3-8.5 (D# group 0 + group 1; groups 2/3 zero for 2D).
__device__ inline void tdm_load_2d_bf16(unsigned lds_byte_addr,
                                        const unsigned short* gtile,
                                        unsigned tensor_d0, unsigned tensor_d1,
                                        unsigned long long stride0,
                                        unsigned tile_d0, unsigned tile_d1)
{
    unsigned long long ga = (unsigned long long)(uintptr_t)gtile;
    tdm_g0_t g0;
    g0[0] = 1u;                                   // count=1, user mode
    g0[1] = lds_byte_addr;                        // LDS dest (bytes)
    g0[2] = (unsigned)(ga & 0xffffffffu);         // global addr [31:0]
    g0[3] = (unsigned)((ga >> 32) & 0x1ffffffu)   // global addr [56:32]
          | 0x80000000u;                          // type=2 ("image")
    tdm_g1_t g1;
    g1[0] = (int)((1u << 16)      // data_size = 2 bytes
                | (1u << 20)      // pad_enable
                | (3u << 22)      // pad_interval: 16 DWORDs
                | (3u << 25));    // pad_amount:   4 DWORDs
    g1[1] = (int)((tensor_d0 & 0xffffu) << 16);
    g1[2] = (int)(((tensor_d0 >> 16) & 0xffffu) | ((tensor_d1 & 0xffffu) << 16));
    g1[3] = (int)(((tensor_d1 >> 16) & 0xffffu) | ((tile_d0 & 0xffffu) << 16));
    g1[4] = (int)(tile_d1 & 0xffffu);             // tile_dim1; tile_dim2 = 0
    g1[5] = (int)(stride0 & 0xffffffffull);
    g1[6] = (int)((stride0 >> 32) & 0xffffull);   // stride1 = 0 (2D)
    g1[7] = 0;
    tdm_g2_t gz4 = {0, 0, 0, 0};
    tdm_g1_t gz8 = {0, 0, 0, 0, 0, 0, 0, 0};
    // clang-23 / therock-10.0 signature: (v4u, v8i, v4i, v4i, v8i, i32 cpol)
    __builtin_amdgcn_tensor_load_to_lds(g0, g1, gz4, gz4, gz8, 0);
}

// C[b,n] = sum_k A[b,k] * W[n,k] + bias[n]
// A: [Bdim, Kd] (fp32 or bf16-as-ushort); Wbf: [N, Kd] bf16; Cout fp32 or bf16
template <typename AT, bool OUT_BF16>
__global__ __launch_bounds__(256)
void gemm_bias_wmma(const AT* __restrict__ A,
                    const unsigned short* __restrict__ Wbf,
                    const float* __restrict__ bias,
                    void* __restrict__ Cout,
                    int Bdim, int N, int Kd)
{
    constexpr bool A_IS_F32 = (sizeof(AT) == 4);

    __shared__ unsigned short lds_a[2][TBM * LDA];
    __shared__ unsigned short lds_w[2][TBN * LDW];

    const int tid  = threadIdx.x;
    const int lane = tid & 31;
    const int half = lane >> 4;
    const int lr   = lane & 15;
    const int wv   = __builtin_amdgcn_readfirstlane((int)(threadIdx.x >> 5)); // wave id (SGPR)

    const int colBase = blockIdx.x * TBN;   // N tiles fastest -> A band stays in L2
    const int rowBase = blockIdx.y * TBM;

    const int wr = (tid >> 5) >> 1;   // 0..3 : 32-row group
    const int wc = (tid >> 5) & 1;    // 0..1 : 32-col group

    // ---- A staging registers (double-buffer pipeline) ----
    float4 fa[4];   // AT == float   : 16 elems/thread/step
    u16x8  ha[2];   // AT == ushort  : 16 elems/thread/step

    auto loadA = [&](int k0) {
        if constexpr (A_IS_F32) {
            #pragma unroll
            for (int it = 0; it < 4; ++it) {
                int idx = it * 256 + tid;
                int row = idx >> 3, ch = idx & 7;          // 8 float4 chunks per row
                fa[it] = *reinterpret_cast<const float4*>(
                    A + (size_t)(rowBase + row) * Kd + k0 + ch * 4);
            }
        } else {
            #pragma unroll
            for (int it = 0; it < 2; ++it) {
                int idx = it * 256 + tid;
                int row = idx >> 1, h8 = idx & 1;          // 2 ushort8 chunks per row
                ha[it] = *reinterpret_cast<const u16x8*>(
                    A + (size_t)(rowBase + row) * Kd + k0 + h8 * 8);
            }
        }
    };
    auto storeA = [&](unsigned short* dst) {
        if constexpr (A_IS_F32) {
            #pragma unroll
            for (int it = 0; it < 4; ++it) {
                int idx = it * 256 + tid;
                int row = idx >> 3, ch = idx & 7;
                uint2 p;
                p.x = pack_bf16x2(fa[it].x, fa[it].y);
                p.y = pack_bf16x2(fa[it].z, fa[it].w);
                *reinterpret_cast<uint2*>(dst + row * LDA + ch * 4) = p;   // ds_store_b64
            }
        } else {
            #pragma unroll
            for (int it = 0; it < 2; ++it) {
                int idx = it * 256 + tid;
                int row = idx >> 1, h8 = idx & 1;
                *reinterpret_cast<u16x8*>(dst + row * LDA + h8 * 8) = ha[it]; // ds_store_b128
            }
        }
    };
    // Each wave DMAs its own 8-row slice of the W tile (own TENSORcnt -> per-wave wait).
    auto tdmW = [&](int buf, int k0) {
        const unsigned short* g = Wbf + ((size_t)(colBase + wv * 8) * Kd + k0);
        unsigned laddr = (unsigned)(uintptr_t)&lds_w[buf][wv * 8 * LDW];
        tdm_load_2d_bf16(laddr, g, (unsigned)Kd, (unsigned)N,
                         (unsigned long long)Kd, (unsigned)TK, 8u);
    };

    v8f acc[2][2];
    #pragma unroll
    for (int i = 0; i < 2; ++i)
        #pragma unroll
        for (int j = 0; j < 2; ++j)
            acc[i][j] = {};

    const int nsteps = Kd / TK;

    // ---- prologue: fill buffer 0 ----
    loadA(0);
    storeA(lds_a[0]);
    tdmW(0, 0);
    __builtin_amdgcn_s_wait_tensorcnt(0);
    __syncthreads();

    for (int s = 0; s < nsteps; ++s) {
        const int  cur  = s & 1, nxt = cur ^ 1;
        const bool more = (s + 1) < nsteps;

        // issue next tile's traffic before the math (overlap with WMMA pipe)
        if (more) {
            loadA((s + 1) * TK);
            tdmW(nxt, (s + 1) * TK);
        }
        // warm L2 for the A stream two steps ahead (global_prefetch_b8)
        if (s + 2 < nsteps) {
            int row = tid >> 3;
            __builtin_prefetch(A + (size_t)(rowBase + row) * Kd + (s + 2) * TK, 0, 3);
        }

        // ---- fragments from current buffers (ISA 7.12.2 layouts) ----
        Frag afr[2], bfr[2];
        #pragma unroll
        for (int i = 0; i < 2; ++i) {
            int row = wr * 32 + i * 16 + lr;
            const unsigned short* base = &lds_a[cur][row * LDA];
            afr[i].h[0] = *reinterpret_cast<const u16x8*>(base + (half ? 8 : 0));
            afr[i].h[1] = *reinterpret_cast<const u16x8*>(base + 16 + (half ? 8 : 0));
        }
        #pragma unroll
        for (int j = 0; j < 2; ++j) {
            int n = wc * 32 + j * 16 + lr;
            const unsigned short* base = &lds_w[cur][n * LDW + half * 8];
            bfr[j].h[0] = *reinterpret_cast<const u16x8*>(base);
            bfr[j].h[1] = *reinterpret_cast<const u16x8*>(base + 16);
        }

        #pragma unroll
        for (int i = 0; i < 2; ++i)
            #pragma unroll
            for (int j = 0; j < 2; ++j)
                acc[i][j] = __builtin_amdgcn_wmma_f32_16x16x32_bf16(
                    false, afr[i].bf, false, bfr[j].bf,
                    (short)0, acc[i][j], false, false);

        if (more) {
            storeA(lds_a[nxt]);                    // waits the global loads, fills next buf
            __builtin_amdgcn_s_wait_tensorcnt(0);  // next W tile resident
        }
        if (more) __syncthreads();
    }

    // ---- bias + store ----
    #pragma unroll
    for (int j = 0; j < 2; ++j) {
        int n    = colBase + wc * 32 + j * 16 + lr;
        float bv = bias[n];
        #pragma unroll
        for (int i = 0; i < 2; ++i) {
            int m0 = rowBase + wr * 32 + i * 16 + half * 8;
            #pragma unroll
            for (int r = 0; r < 8; ++r) {
                float v = acc[i][j][r] + bv;
                size_t off = (size_t)(m0 + r) * N + n;
                if constexpr (OUT_BF16)
                    ((unsigned short*)Cout)[off] = f32_to_bf16(v);
                else
                    ((float*)Cout)[off] = v;
            }
        }
    }
}

// fp32 -> bf16 bulk conversion (vectorized: float4 in, uint2 out)
__global__ __launch_bounds__(256)
void cvt_f32_to_bf16(const float* __restrict__ in, unsigned short* __restrict__ out, int n4)
{
    int i = blockIdx.x * blockDim.x + threadIdx.x;
    if (i < n4) {
        float4 f = reinterpret_cast<const float4*>(in)[i];
        uint2 p;
        p.x = pack_bf16x2(f.x, f.y);
        p.y = pack_bf16x2(f.z, f.w);
        reinterpret_cast<uint2*>(out)[i] = p;
    }
}

extern "C" void kernel_launch(void* const* d_in, const int* in_sizes, int n_in,
                              void* d_out, int out_size, void* d_ws, size_t ws_size,
                              hipStream_t stream)
{
    const float* X  = (const float*)d_in[0];  // [B, M]
    const float* W1 = (const float*)d_in[1];  // [K, M]
    const float* b1 = (const float*)d_in[2];  // [K]
    const float* W2 = (const float*)d_in[3];  // [M, K]
    const float* b2 = (const float*)d_in[4];  // [M]
    float* out = (float*)d_out;               // [B, M]

    const int Mdim = in_sizes[4];             // 2048
    const int Kdim = in_sizes[2];             // 512
    const int Bdim = in_sizes[0] / Mdim;      // 16384

    // workspace layout (bf16 ushorts)
    unsigned short* w1bf   = (unsigned short*)d_ws;               // [K, M]
    unsigned short* w2bf   = w1bf + (size_t)Kdim * Mdim;          // [M, K]
    unsigned short* hidden = w2bf + (size_t)Mdim * Kdim;          // [B, K]

    dim3 blk(256);

    // Pre-convert weights once (tiny: 12 MB of traffic, then L2-resident)
    int n4w = (Kdim * Mdim) / 4;
    cvt_f32_to_bf16<<<dim3((n4w + 255) / 256), blk, 0, stream>>>(W1, w1bf, n4w);
    cvt_f32_to_bf16<<<dim3((n4w + 255) / 256), blk, 0, stream>>>(W2, w2bf, n4w);

    // Layer 1: hidden = X @ W1^T + b1   (A fp32 converted in-flight, out bf16)
    dim3 g1(Kdim / TBN, Bdim / TBM);
    gemm_bias_wmma<float, true><<<g1, blk, 0, stream>>>(
        X, w1bf, b1, hidden, Bdim, Kdim, Mdim);

    // Layer 2: out = hidden @ W2^T + b2 (A bf16, out fp32)
    dim3 g2(Mdim / TBN, Bdim / TBM);
    gemm_bias_wmma<unsigned short, false><<<g2, blk, 0, stream>>>(
        hidden, w2bf, b2, out, Bdim, Mdim, Kdim);
}
